// MomentumGRU_87351044866268
// MI455X (gfx1250) — matrix operand order repaired and checked by reading, not compiled, not run on hardware
//
#include <hip/hip_runtime.h>
#include <hip/hip_bf16.h>
#include <math.h>

// ---------------------------------------------------------------------------
// MomentumGRU (2-layer) for MI455X / gfx1250 — fused cell kernel
//   v_t = mu*v_{t-1} + eps*(inp @ Wx + bx)
//   r,z = sigmoid(v_rz + (h@Wh)_rz); n = tanh(v_n + r*(h@Wh)_n)
//   h   = (1-z)*h + z*n
// Per (layer, step): ONE kernel. Each wave owns 16 rows x 16 gate-cols and
// computes the r/z/n column triple {j, j+H, j+2H} so the whole gate is
// register-resident (no G buffer, no pointwise kernel). h is ping-ponged.
// ---------------------------------------------------------------------------

typedef __attribute__((ext_vector_type(16))) __bf16 v16bf;
typedef __attribute__((ext_vector_type(8)))  __bf16 v8bf;
typedef __attribute__((ext_vector_type(8)))  float  v8f;

#define B_   64
#define T_   512
#define D_   512
#define H_   1024
#define TH_  3072            // 3*H
#define MU_C  0.9f
#define EPS_C 1.0f

// -------------------------- weight transpose+convert ------------------------
// W: fp32 [K, N] row-major  ->  Wt: bf16 [N, K] row-major (i.e. column-major W)
__global__ __launch_bounds__(256)
void transpose_convert_kernel(const float* __restrict__ W, __bf16* __restrict__ Wt,
                              int K, int N)
{
    __shared__ float tile[32][33];
    const int tx = threadIdx.x & 31;
    const int ty = threadIdx.x >> 5;          // 0..7
    const int n0 = blockIdx.x * 32;
    const int k0 = blockIdx.y * 32;
    #pragma unroll
    for (int i = 0; i < 32; i += 8)
        tile[ty + i][tx] = W[(size_t)(k0 + ty + i) * N + (n0 + tx)];
    __syncthreads();
    #pragma unroll
    for (int i = 0; i < 32; i += 8)
        Wt[(size_t)(n0 + ty + i) * K + (k0 + tx)] = (__bf16)tile[tx][ty + i];
}

// ------------------------------ constant fill -------------------------------
__global__ void fill_kernel(float* __restrict__ p, float v, int n)
{
    int i = blockIdx.x * blockDim.x + threadIdx.x;
    if (i < n) p[i] = v;
}

// ------------------------- fused GRU cell (one layer) ------------------------
// grid = (B/16, H/64), block = 128 threads (4 waves).
// Wave w handles rows [r0, r0+16) x gate-cols [jb + 16w, jb + 16w + 16),
// i.e. three 16x16 WMMA tiles at columns j, j+H, j+2H of the 3H-wide GEMMs.
__global__ __launch_bounds__(128)
void fused_cell_kernel(const float* __restrict__ inp, long ldInp, int K1,
                       const __bf16* __restrict__ WxT, const float* __restrict__ bx,
                       const float* __restrict__ hRead, const __bf16* __restrict__ WhT,
                       float* __restrict__ v, float* __restrict__ hWrite,
                       float* __restrict__ outT, long outStrideB)
{
    __shared__ __bf16 At[16][1040];            // 16 x K panel (K<=1024), 2080B pitch

    const int tid  = threadIdx.x;
    const int lane = tid & 31;                 // wave32
    const int wid  = tid >> 5;                 // 0..3
    const int l    = lane & 15;
    const int hi   = lane >> 4;                // half-wave K split

    const int r0 = blockIdx.x * 16;            // batch-row strip
    const int jw = blockIdx.y * 64 + wid * 16; // gate-col base for this wave

    // ---- stage a 16 x K fp32 panel as bf16 into LDS (vectorized) -----------
    auto stage = [&](const float* __restrict__ src, long lda, int K) {
        const int lg = (K == 512) ? 6 : 7;     // log2(K/8)
        const int nseg = K >> 3;
        for (int e = tid; e < 16 * nseg; e += 128) {
            const int r = e >> lg;
            const int s = e & (nseg - 1);
            const float* p = src + (long)(r0 + r) * lda + s * 8;
            const float4 f0 = *(const float4*)(p);
            const float4 f1 = *(const float4*)(p + 4);
            v8bf pk;
            pk[0] = (__bf16)f0.x; pk[1] = (__bf16)f0.y;
            pk[2] = (__bf16)f0.z; pk[3] = (__bf16)f0.w;
            pk[4] = (__bf16)f1.x; pk[5] = (__bf16)f1.y;
            pk[6] = (__bf16)f1.z; pk[7] = (__bf16)f1.w;
            *(v8bf*)&At[r][s * 8] = pk;        // ds_store_b128
        }
    };

    // ---- K-loop: 3 gate tiles share one A fragment -------------------------
    // Hoisted, incremented B pointers -> immediate-offset loads after unroll.
    // Unconditional speculative prefetch (dropped past end; never faults).
    auto kloop = [&](int K, const __bf16* __restrict__ Wt, v8f* acc) {
        const __bf16* arow = &At[l][hi * 16];                      // LDS A row base
        const __bf16* bp0  = Wt + (size_t)(jw + l         ) * K + hi * 16;
        const __bf16* bp1  = Wt + (size_t)(jw + l +     H_) * K + hi * 16;
        const __bf16* bp2  = Wt + (size_t)(jw + l + 2 * H_) * K + hi * 16;
        for (int k0 = 0; k0 < K; k0 += 32) {
            const v8bf alo = *(const v8bf*)(arow + k0);            // ds_load_b128
            const v8bf ahi = *(const v8bf*)(arow + k0 + 8);
            v16bf af;
            #pragma unroll
            for (int i = 0; i < 8; ++i) { af[i] = alo[i]; af[i + 8] = ahi[i]; }

            __builtin_prefetch(bp0 + 32, 0, 3);                    // global_prefetch_b8
            __builtin_prefetch(bp1 + 32, 0, 3);
            __builtin_prefetch(bp2 + 32, 0, 3);

            const v16bf b0 = *(const v16bf*)bp0;                   // 2x global_load_b128
            const v16bf b1 = *(const v16bf*)bp1;
            const v16bf b2 = *(const v16bf*)bp2;

            acc[0] = __builtin_amdgcn_wmma_f32_16x16x32_bf16(
                         false, af, false, b0, (short)0, acc[0], true,  false);
            acc[1] = __builtin_amdgcn_wmma_f32_16x16x32_bf16(
                         false, af, false, b1, (short)0, acc[1], true,  false);
            acc[2] = __builtin_amdgcn_wmma_f32_16x16x32_bf16(
                         false, af, false, b2, (short)0, acc[2], false, false);

            bp0 += 32; bp1 += 32; bp2 += 32;
        }
    };

    // ================= phase 0: v = mu*v + eps*(inp@Wx + bx) ================
    stage(inp, ldInp, K1);
    __syncthreads();
    v8f accv[3] = {};
    kloop(K1, WxT, accv);

    #pragma unroll
    for (int g = 0; g < 3; ++g) {
        const int c = jw + l + g * H_;
        const float bxc = bx[c];
        #pragma unroll
        for (int gv = 0; gv < 8; ++gv) {
            const int row = r0 + hi * 8 + gv;
            const size_t vi = (size_t)row * TH_ + c;
            const float vn = MU_C * v[vi] + EPS_C * (accv[g][gv] + bxc);
            v[vi] = vn;
            accv[g][gv] = vn;                  // keep for gating
        }
    }

    // ================= phase 1: acch = h @ Wh ===============================
    __syncthreads();                           // all waves done reading phase-0 LDS
    stage(hRead, (long)H_, H_);
    __syncthreads();
    v8f acch[3] = {};
    kloop(H_, WhT, acch);

    // ================= gates: all register-resident =========================
    #pragma unroll
    for (int gv = 0; gv < 8; ++gv) {
        const int row = r0 + hi * 8 + gv;
        const int j   = jw + l;
        const float r = 1.0f / (1.0f + __expf(-(accv[0][gv] + acch[0][gv])));
        const float z = 1.0f / (1.0f + __expf(-(accv[1][gv] + acch[1][gv])));
        const float n = tanhf(accv[2][gv] + r * acch[2][gv]);
        const float hold = hRead[(size_t)row * H_ + j];
        const float hnew = (1.0f - z) * hold + z * n;
        hWrite[(size_t)row * H_ + j] = hnew;
        if (outT) outT[(size_t)row * outStrideB + j] = hnew;
    }
}

// --------------------------------- launcher ---------------------------------
extern "C" void kernel_launch(void* const* d_in, const int* in_sizes, int n_in,
                              void* d_out, int out_size, void* d_ws, size_t ws_size,
                              hipStream_t stream)
{
    (void)in_sizes; (void)n_in; (void)out_size; (void)ws_size;

    const float* x   = (const float*)d_in[0];   // [B,T,D]
    const float* Wx0 = (const float*)d_in[1];   // [D,3H]
    const float* bx0 = (const float*)d_in[2];   // [3H]
    const float* Wh0 = (const float*)d_in[3];   // [H,3H]
    const float* Wx1 = (const float*)d_in[4];   // [H,3H]
    const float* bx1 = (const float*)d_in[5];   // [3H]
    const float* Wh1 = (const float*)d_in[6];   // [H,3H]

    // d_out tuple layout (floats): outputs | h0 | v0 | h1 | v1 | betas
    float* out   = (float*)d_out;
    float* h0A   = out  + (size_t)B_ * T_ * H_;   // final-state slot (odd-parity target)
    float* v0    = h0A  + (size_t)B_ * H_;
    float* h1A   = v0   + (size_t)B_ * TH_;
    float* v1    = h1A  + (size_t)B_ * H_;
    float* betas = v1   + (size_t)B_ * TH_;

    // workspace: transposed bf16 weights + h ping buffers
    char* ws = (char*)d_ws;
    __bf16* WxT0 = (__bf16*)(ws);                                    // [3H, D]
    __bf16* WhT0 = (__bf16*)(ws + (size_t)TH_ * D_ * 2);             // [3H, H]
    __bf16* WxT1 = (__bf16*)(ws + (size_t)TH_ * (D_ + H_) * 2);      // [3H, H]
    __bf16* WhT1 = (__bf16*)(ws + (size_t)TH_ * (D_ + 2 * H_) * 2);  // [3H, H]
    float*  h0B  = (float*) (ws + (size_t)TH_ * (D_ + 3 * H_) * 2);  // [B, H]
    float*  h1B  = h0B + (size_t)B_ * H_;                            // [B, H]

    // zero the recurrent state (h0|v0|h1|v1 contiguous in d_out) each call
    (void)hipMemsetAsync(h0A, 0, (size_t)(2 * B_ * H_ + 2 * B_ * TH_) * sizeof(float), stream);

    // betas = mu everywhere
    {
        const int n = B_ * T_ * 2;
        fill_kernel<<<(n + 255) / 256, 256, 0, stream>>>(betas, MU_C, n);
    }

    // one-time weight transpose+convert (bf16 weights then live in L2: ~22MB)
    transpose_convert_kernel<<<dim3(TH_ / 32, D_ / 32), 256, 0, stream>>>(Wx0, WxT0, D_, TH_);
    transpose_convert_kernel<<<dim3(TH_ / 32, H_ / 32), 256, 0, stream>>>(Wh0, WhT0, H_, TH_);
    transpose_convert_kernel<<<dim3(TH_ / 32, H_ / 32), 256, 0, stream>>>(Wx1, WxT1, H_, TH_);
    transpose_convert_kernel<<<dim3(TH_ / 32, H_ / 32), 256, 0, stream>>>(Wh1, WhT1, H_, TH_);

    const dim3 grid(B_ / 16, H_ / 64);         // 64 blocks of 4 waves
    const long xRowStride = (long)T_ * D_;     // x[b,t,:] row stride over b
    const long outStrideB = (long)T_ * H_;

    for (int t = 0; t < T_; ++t) {
        // ping-pong so in-kernel h writes never race other blocks' h reads;
        // odd-parity writes land in the d_out slot, so t=511 leaves final h there.
        float* h0r = (t & 1) ? h0B : h0A;
        float* h0w = (t & 1) ? h0A : h0B;
        float* h1r = (t & 1) ? h1B : h1A;
        float* h1w = (t & 1) ? h1A : h1B;

        // ---- layer 0 ----
        fused_cell_kernel<<<grid, 128, 0, stream>>>(
            x + (size_t)t * D_, xRowStride, D_, WxT0, bx0,
            h0r, WhT0, v0, h0w, (float*)nullptr, 0);

        // ---- layer 1 (input = freshly written h0 of this step) ----
        fused_cell_kernel<<<grid, 128, 0, stream>>>(
            h0w, (long)H_, H_, WxT1, bx1,
            h1r, WhT1, v1, h1w, out + (size_t)t * H_, outStrideB);
    }
}
